// SwinTransformerBlock_81595788689890
// MI455X (gfx1250) — matrix operand order repaired and checked by reading, not compiled, not run on hardware
//
#include <hip/hip_runtime.h>
#include <hip/hip_bf16.h>

typedef __bf16 bf16;
typedef __attribute__((ext_vector_type(8)))  bf16  v8bf;
typedef __attribute__((ext_vector_type(16))) bf16  v16bf;
typedef __attribute__((ext_vector_type(8)))  float v8f;

// ---------------------------------------------------------------------------
// WMMA helpers (gfx1250, wave32).  D = A(16x32 bf16) * B(32x16 bf16) + C(f32)
// ---------------------------------------------------------------------------
static __device__ inline v8f wmma_bf16(v16bf a, v16bf b, v8f c) {
  return __builtin_amdgcn_wmma_f32_16x16x32_bf16(false, a, false, b, (short)0, c,
                                                 false, false);
}

// A fragment: A is row-major [M, lda], tile origin (m0, k0), K-slab of 32.
// Lane layout (ISA 7.12.2): lanes 0-15 m=0..15 hold K {0..7,16..23};
// lanes 16-31 hold K {8..15,24..31}.
static __device__ inline v16bf load_a_frag(const bf16* __restrict__ A, int lda,
                                           int m0, int k0) {
  int lane = threadIdx.x & 31;
  const bf16* p = A + (size_t)(m0 + (lane & 15)) * lda + (k0 + ((lane >> 4) << 3));
  v8bf lo = *(const v8bf*)p;         // K = base8 + 0..7
  v8bf hi = *(const v8bf*)(p + 16);  // K = base8 + 16..23
  v16bf a;
#pragma unroll
  for (int i = 0; i < 8; ++i) { a[i] = lo[i]; a[i + 8] = hi[i]; }
  return a;
}

// B fragment from weight stored [N, ldk] row-major (i.e. W[n][k], the natural
// layout of "out = x @ W.T" weights).  Lane n = lane%16, K =
// (lane>=16 ? 8 : 0) + {0..7,16..23} (same as A) -> two 16B loads.
static __device__ inline v16bf load_b_frag(const bf16* __restrict__ W, int ldk,
                                           int n0, int k0) {
  int lane = threadIdx.x & 31;
  const bf16* p = W + (size_t)(n0 + (lane & 15)) * ldk + (k0 + ((lane >> 4) << 3));
  v8bf lo = *(const v8bf*)p;
  v8bf hi = *(const v8bf*)(p + 16);
  v16bf b;
#pragma unroll
  for (int i = 0; i < 8; ++i) { b[i] = lo[i]; b[i + 8] = hi[i]; }
  return b;
}

// One wave computes a (MT*16) x (NT*16) tile; each B fragment is reused
// across the MT m-sub-tiles to raise WMMA:VMEM issue ratio.
template <int MT, int NT>
static __device__ inline void gemm_tile(const bf16* __restrict__ A, int lda,
                                        const bf16* __restrict__ W, int ldk,
                                        int m0, int n0, int K, v8f acc[MT][NT]) {
  for (int k0 = 0; k0 < K; k0 += 32) {
    v16bf a[MT];
#pragma unroll
    for (int i = 0; i < MT; ++i) a[i] = load_a_frag(A, lda, m0 + 16 * i, k0);
#pragma unroll
    for (int j = 0; j < NT; ++j) {
      v16bf b = load_b_frag(W, ldk, n0 + 16 * j, k0);
#pragma unroll
      for (int i = 0; i < MT; ++i) acc[i][j] = wmma_bf16(a[i], b, acc[i][j]);
    }
  }
}

static __device__ inline float xor_sum(float v, int width) {
#pragma unroll
  for (int off = 1; off < 16; off <<= 1) v += __shfl_xor(v, off, 16);
  if (width == 32) v += __shfl_xor(v, 16, 32);
  return v;
}

// ---------------------------------------------------------------------------
// Problem constants
// ---------------------------------------------------------------------------
#define BSZ   32
#define HW    56
#define CCH   128
#define HID   512
#define WSW   7
#define NWIN  2048      // 32 * 8 * 8
#define NTOK  100352    // NWIN * 49
#define NPAD  64
#define HDIM  32
#define NHEAD 4
#define SHIFT 3

// fp32 -> bf16 convert
__global__ void cvt_kernel(const float* __restrict__ in, bf16* __restrict__ out, int n) {
  int i = blockIdx.x * 256 + threadIdx.x;
  if (i < n) out[i] = (bf16)in[i];
}

// LayerNorm1 + roll(-3,-3) + window partition.  One wave per output row.
__global__ __launch_bounds__(256) void ln1_kernel(const float* __restrict__ x,
                                                  const float* __restrict__ g,
                                                  const float* __restrict__ bb,
                                                  bf16* __restrict__ hb) {
  int row = blockIdx.x * 8 + (threadIdx.x >> 5);
  int lane = threadIdx.x & 31;
  int win = row / 49, t = row - win * 49;
  int b = win >> 6, wc = win & 63, wi = wc >> 3, wj = wc & 7;
  int ti = t / WSW, tj = t - ti * WSW;
  int y = wi * WSW + ti + SHIFT; if (y >= HW) y -= HW;
  int z = wj * WSW + tj + SHIFT; if (z >= HW) z -= HW;
  const float* xr = x + ((size_t)b * (HW * HW) + y * HW + z) * CCH;
  float4 v = *(const float4*)(xr + lane * 4);
  float s  = v.x + v.y + v.z + v.w;
  float s2 = v.x * v.x + v.y * v.y + v.z * v.z + v.w * v.w;
  s = xor_sum(s, 32);  s2 = xor_sum(s2, 32);
  float mean = s * (1.f / CCH);
  float var  = s2 * (1.f / CCH) - mean * mean;
  float rs = rsqrtf(var + 1e-5f);
  int c = lane * 4;
  bf16* o = hb + (size_t)row * CCH + c;
  o[0] = (bf16)((v.x - mean) * rs * g[c + 0] + bb[c + 0]);
  o[1] = (bf16)((v.y - mean) * rs * g[c + 1] + bb[c + 1]);
  o[2] = (bf16)((v.z - mean) * rs * g[c + 2] + bb[c + 2]);
  o[3] = (bf16)((v.w - mean) * rs * g[c + 3] + bb[c + 3]);
}

// LayerNorm2 over x2 (natural token order) -> bf16
__global__ __launch_bounds__(256) void ln2_kernel(const float* __restrict__ x2,
                                                  const float* __restrict__ g,
                                                  const float* __restrict__ bb,
                                                  bf16* __restrict__ h2) {
  int row = blockIdx.x * 8 + (threadIdx.x >> 5);
  int lane = threadIdx.x & 31;
  const float* xr = x2 + (size_t)row * CCH;
  float4 v = *(const float4*)(xr + lane * 4);
  float s  = v.x + v.y + v.z + v.w;
  float s2 = v.x * v.x + v.y * v.y + v.z * v.z + v.w * v.w;
  s = xor_sum(s, 32);  s2 = xor_sum(s2, 32);
  float mean = s * (1.f / CCH);
  float var  = s2 * (1.f / CCH) - mean * mean;
  float rs = rsqrtf(var + 1e-5f);
  int c = lane * 4;
  bf16* o = h2 + (size_t)row * CCH + c;
  o[0] = (bf16)((v.x - mean) * rs * g[c + 0] + bb[c + 0]);
  o[1] = (bf16)((v.y - mean) * rs * g[c + 1] + bb[c + 1]);
  o[2] = (bf16)((v.z - mean) * rs * g[c + 2] + bb[c + 2]);
  o[3] = (bf16)((v.w - mean) * rs * g[c + 3] + bb[c + 3]);
}

// QKV GEMM: [NTOK,128] x [128,384] ; scatter into q/k [win,h,64,32] (q scaled)
// and v transposed [win,h,32,64].
__global__ __launch_bounds__(256) void qkv_kernel(const bf16* __restrict__ hb,
                                                  const bf16* __restrict__ wq,
                                                  const float* __restrict__ qb_bias,
                                                  bf16* __restrict__ q,
                                                  bf16* __restrict__ k,
                                                  bf16* __restrict__ v) {
  int m0 = (blockIdx.x * 8 + (threadIdx.x >> 5)) * 32;
  int n0 = blockIdx.y * 64;
  v8f z = {0, 0, 0, 0, 0, 0, 0, 0};
  v8f acc[2][4] = {{z, z, z, z}, {z, z, z, z}};
  gemm_tile<2, 4>(hb, CCH, wq, CCH, m0, n0, CCH, acc);
  int lane = threadIdx.x & 31;
  int nloc = lane & 15;
  int mof = ((lane >> 4) & 1) * 8;
#pragma unroll
  for (int j = 0; j < 4; ++j) {
    int n = n0 + j * 16 + nloc;
    float bias = qb_bias[n];
    int s = n >> 7, hh = (n >> 5) & 3, d = n & 31;
#pragma unroll
    for (int i = 0; i < 2; ++i) {
#pragma unroll
      for (int r = 0; r < 8; ++r) {
        int m = m0 + i * 16 + mof + r;
        int win = m / 49, t = m - win * 49;
        float val = acc[i][j][r] + bias;
        size_t base = (size_t)(win * NHEAD + hh);
        if (s == 0)      q[(base * NPAD + t) * HDIM + d] = (bf16)(val * 0.17677669529663687f);
        else if (s == 1) k[(base * NPAD + t) * HDIM + d] = (bf16)val;
        else             v[(base * HDIM + d) * NPAD + t] = (bf16)val;
      }
    }
  }
}

// Attention: one wave per (window, head).
__global__ __launch_bounds__(32) void attn_kernel(const bf16* __restrict__ qb,
                                                  const bf16* __restrict__ kb,
                                                  const bf16* __restrict__ vb,
                                                  const float* __restrict__ btab,
                                                  const int* __restrict__ ridx,
                                                  bf16* __restrict__ ob) {
  __shared__ bf16 Pl[NPAD * NPAD];
  int wh = blockIdx.x;
  int win = wh >> 2, h = wh & 3;
  const bf16* q  = qb + (size_t)wh * NPAD * HDIM;
  const bf16* kk = kb + (size_t)wh * NPAD * HDIM;
  const bf16* vt = vb + (size_t)wh * HDIM * NPAD;
  int lane = threadIdx.x;
  int nloc = lane & 15;
  int mof = ((lane >> 4) & 1) * 8;

  v8f z = {0, 0, 0, 0, 0, 0, 0, 0};
  v8f acc[16];
#pragma unroll
  for (int i = 0; i < 16; ++i) acc[i] = z;

  // S = q @ k^T   (K = hd = 32, single WMMA per tile)
#pragma unroll
  for (int mt = 0; mt < 4; ++mt) {
    v16bf a = load_a_frag(q, HDIM, mt * 16, 0);
#pragma unroll
    for (int nt = 0; nt < 4; ++nt) {
      v16bf b = load_b_frag(kk, HDIM, nt * 16, 0);
      acc[mt * 4 + nt] = wmma_bf16(a, b, acc[mt * 4 + nt]);
    }
  }

  // relative-position bias + masked softmax over valid 49 columns
#pragma unroll
  for (int mt = 0; mt < 4; ++mt) {
#pragma unroll
    for (int r = 0; r < 8; ++r) {
      int m = mt * 16 + mof + r;
      float e[4];
      float mx = -3.0e38f;
#pragma unroll
      for (int nt = 0; nt < 4; ++nt) {
        int n = nt * 16 + nloc;
        float s = -3.0e38f;
        if (m < 49 && n < 49)
          s = acc[mt * 4 + nt][r] + btab[ridx[m * 49 + n] * NHEAD + h];
        e[nt] = s;
        mx = fmaxf(mx, s);
      }
#pragma unroll
      for (int off = 1; off < 16; off <<= 1) mx = fmaxf(mx, __shfl_xor(mx, off, 16));
      float sum = 0.f;
#pragma unroll
      for (int nt = 0; nt < 4; ++nt) {
        float p = (e[nt] > -1.0e37f) ? __expf(e[nt] - mx) : 0.f;
        e[nt] = p;
        sum += p;
      }
#pragma unroll
      for (int off = 1; off < 16; off <<= 1) sum += __shfl_xor(sum, off, 16);
      float inv = (sum > 0.f) ? 1.f / sum : 0.f;
#pragma unroll
      for (int nt = 0; nt < 4; ++nt)
        Pl[m * NPAD + nt * 16 + nloc] = (bf16)(e[nt] * inv);
    }
  }
  __syncthreads();

  // O = P @ V  (vt stored [d][j] so it is the row-major [N,K] B operand)
#pragma unroll
  for (int mt = 0; mt < 4; ++mt) {
#pragma unroll
    for (int dt = 0; dt < 2; ++dt) {
      v8f c = z;
#pragma unroll
      for (int k0 = 0; k0 < NPAD; k0 += 32) {
        v16bf a = load_a_frag(Pl, NPAD, mt * 16, k0);
        v16bf b = load_b_frag(vt, NPAD, dt * 16, k0);
        c = wmma_bf16(a, b, c);
      }
#pragma unroll
      for (int r = 0; r < 8; ++r) {
        int m = mt * 16 + mof + r;
        if (m < 49) {
          int d = dt * 16 + nloc;
          ob[((size_t)(win * 49 + m)) * CCH + h * HDIM + d] = (bf16)c[r];
        }
      }
    }
  }
}

// proj GEMM + window reverse + un-roll + residual -> x2 (fp32, in d_out)
__global__ __launch_bounds__(256) void proj_kernel(const bf16* __restrict__ ob,
                                                   const bf16* __restrict__ wp,
                                                   const float* __restrict__ pb,
                                                   const float* __restrict__ x,
                                                   float* __restrict__ out) {
  int m0 = (blockIdx.x * 8 + (threadIdx.x >> 5)) * 32;
  int n0 = blockIdx.y * 64;
  v8f z = {0, 0, 0, 0, 0, 0, 0, 0};
  v8f acc[2][4] = {{z, z, z, z}, {z, z, z, z}};
  gemm_tile<2, 4>(ob, CCH, wp, CCH, m0, n0, CCH, acc);
  int lane = threadIdx.x & 31;
  int nloc = lane & 15;
  int mof = ((lane >> 4) & 1) * 8;
#pragma unroll
  for (int j = 0; j < 4; ++j) {
    int n = n0 + j * 16 + nloc;
    float bias = pb[n];
#pragma unroll
    for (int i = 0; i < 2; ++i) {
#pragma unroll
      for (int r = 0; r < 8; ++r) {
        int m = m0 + i * 16 + mof + r;
        int win = m / 49, t = m - win * 49;
        int b = win >> 6, wc = win & 63, wi = wc >> 3, wj = wc & 7;
        int ti = t / WSW, tj = t - ti * WSW;
        int y = wi * WSW + ti + SHIFT; if (y >= HW) y -= HW;
        int zz = wj * WSW + tj + SHIFT; if (zz >= HW) zz -= HW;
        size_t idx = ((size_t)b * (HW * HW) + y * HW + zz) * CCH + n;
        out[idx] = acc[i][j][r] + bias + x[idx];
      }
    }
  }
}

// MLP fc1 + exact GELU
__global__ __launch_bounds__(256) void mlp1_kernel(const bf16* __restrict__ h2,
                                                   const bf16* __restrict__ w1,
                                                   const float* __restrict__ b1,
                                                   bf16* __restrict__ h1) {
  int m0 = (blockIdx.x * 8 + (threadIdx.x >> 5)) * 32;
  int n0 = blockIdx.y * 64;
  v8f z = {0, 0, 0, 0, 0, 0, 0, 0};
  v8f acc[2][4] = {{z, z, z, z}, {z, z, z, z}};
  gemm_tile<2, 4>(h2, CCH, w1, CCH, m0, n0, CCH, acc);
  int lane = threadIdx.x & 31;
  int nloc = lane & 15;
  int mof = ((lane >> 4) & 1) * 8;
#pragma unroll
  for (int j = 0; j < 4; ++j) {
    int n = n0 + j * 16 + nloc;
    float bias = b1[n];
#pragma unroll
    for (int i = 0; i < 2; ++i) {
#pragma unroll
      for (int r = 0; r < 8; ++r) {
        int m = m0 + i * 16 + mof + r;
        float t = acc[i][j][r] + bias;
        t = 0.5f * t * (1.f + erff(t * 0.70710678118654752f));
        h1[(size_t)m * HID + n] = (bf16)t;
      }
    }
  }
}

// MLP fc2 + residual into d_out
__global__ __launch_bounds__(256) void mlp2_kernel(const bf16* __restrict__ h1,
                                                   const bf16* __restrict__ w2,
                                                   const float* __restrict__ b2,
                                                   float* __restrict__ out) {
  int m0 = (blockIdx.x * 8 + (threadIdx.x >> 5)) * 32;
  int n0 = blockIdx.y * 64;
  v8f z = {0, 0, 0, 0, 0, 0, 0, 0};
  v8f acc[2][4] = {{z, z, z, z}, {z, z, z, z}};
  gemm_tile<2, 4>(h1, HID, w2, HID, m0, n0, HID, acc);
  int lane = threadIdx.x & 31;
  int nloc = lane & 15;
  int mof = ((lane >> 4) & 1) * 8;
#pragma unroll
  for (int j = 0; j < 4; ++j) {
    int n = n0 + j * 16 + nloc;
    float bias = b2[n];
#pragma unroll
    for (int i = 0; i < 2; ++i) {
#pragma unroll
      for (int r = 0; r < 8; ++r) {
        int m = m0 + i * 16 + mof + r;
        size_t idx = (size_t)m * CCH + n;
        out[idx] = acc[i][j][r] + bias + out[idx];
      }
    }
  }
}

// ---------------------------------------------------------------------------
extern "C" void kernel_launch(void* const* d_in, const int* in_sizes, int n_in,
                              void* d_out, int out_size, void* d_ws, size_t ws_size,
                              hipStream_t stream) {
  const float* x    = (const float*)d_in[0];
  const float* n1g  = (const float*)d_in[3];
  const float* n1b  = (const float*)d_in[4];
  const float* qkvw = (const float*)d_in[5];
  const float* qkvb = (const float*)d_in[6];
  const float* btab = (const float*)d_in[7];
  const int*   ridx = (const int*)d_in[8];
  const float* pw   = (const float*)d_in[9];
  const float* pb   = (const float*)d_in[10];
  const float* n2g  = (const float*)d_in[11];
  const float* n2b  = (const float*)d_in[12];
  const float* w1f  = (const float*)d_in[13];
  const float* b1   = (const float*)d_in[14];
  const float* w2f  = (const float*)d_in[15];
  const float* b2   = (const float*)d_in[16];
  float* out = (float*)d_out;
  char* ws = (char*)d_ws;

  bf16* WQ = (bf16*)(ws);                    // 384*128
  bf16* WP = (bf16*)(ws + 98304);            // 128*128
  bf16* W1 = (bf16*)(ws + 131072);           // 512*128
  bf16* W2 = (bf16*)(ws + 262144);           // 128*512
  bf16* HB = (bf16*)(ws + 393216);           // [NTOK,128] bf16
  bf16* QB = (bf16*)(ws + 26083328);         // [2048,4,64,32]
  bf16* KB = (bf16*)(ws + 59637760);
  bf16* VB = (bf16*)(ws + 93192192);         // transposed [2048,4,32,64]
  bf16* OB = (bf16*)(ws + 126746624);        // [NTOK,128]
  bf16* H1 = HB;                             // alias (hb/q/k/v dead by mlp1)
  bf16* H2 = OB;                             // alias (ob dead after proj)

  cvt_kernel<<<192, 256, 0, stream>>>(qkvw, WQ, 49152);
  cvt_kernel<<<64, 256, 0, stream>>>(pw, WP, 16384);
  cvt_kernel<<<256, 256, 0, stream>>>(w1f, W1, 65536);
  cvt_kernel<<<256, 256, 0, stream>>>(w2f, W2, 65536);

  ln1_kernel<<<12544, 256, 0, stream>>>(x, n1g, n1b, HB);

  hipMemsetAsync(QB, 0, (size_t)3 * 33554432, stream);  // zero q/k/v pads

  qkv_kernel<<<dim3(392, 6), 256, 0, stream>>>(HB, WQ, qkvb, QB, KB, VB);
  attn_kernel<<<8192, 32, 0, stream>>>(QB, KB, VB, btab, ridx, OB);
  proj_kernel<<<dim3(392, 2), 256, 0, stream>>>(OB, WP, pb, x, out);
  ln2_kernel<<<12544, 256, 0, stream>>>(out, n2g, n2b, H2);
  mlp1_kernel<<<dim3(392, 8), 256, 0, stream>>>(H2, W1, b1, H1);
  mlp2_kernel<<<dim3(392, 2), 256, 0, stream>>>(H1, W2, b2, out);
}
